// ScaledDotProductAttention_69191923139054
// MI455X (gfx1250) — matrix operand order repaired and checked by reading, not compiled, not run on hardware
//
#include <hip/hip_runtime.h>

#define S_LEN 8192
#define D_LEN 1024

typedef __bf16 bf16;
typedef bf16  bf16x4  __attribute__((ext_vector_type(4)));
typedef bf16  bf16x8  __attribute__((ext_vector_type(8)));
typedef bf16  bf16x16 __attribute__((ext_vector_type(16)));
typedef float v8f     __attribute__((ext_vector_type(8)));
typedef unsigned int u32x4 __attribute__((ext_vector_type(4)));
typedef unsigned int u32x8 __attribute__((ext_vector_type(8)));

// ---------------- pre-pass: fp32 -> bf16 (row-major) ----------------
__global__ void cvt_bf16_kernel(const float* __restrict__ src,
                                bf16* __restrict__ dst, int n) {
  int i = (blockIdx.x * blockDim.x + threadIdx.x) * 4;
  if (i + 3 < n) {
    float4 f = *(const float4*)(src + i);
    bf16x4 o;
    o[0] = (bf16)f.x; o[1] = (bf16)f.y; o[2] = (bf16)f.z; o[3] = (bf16)f.w;
    *(bf16x4*)(dst + i) = o;
  }
}

// ---------------- pre-pass: fp32 [S][D] -> bf16 transposed [D][S] ----------------
__global__ void cvt_transpose_kernel(const float* __restrict__ src,
                                     bf16* __restrict__ dst) {
  size_t i = (size_t)blockIdx.x * blockDim.x + threadIdx.x; // output idx = d*S + s
  int d = (int)(i >> 13);     // / 8192
  int s = (int)(i & 8191);
  dst[i] = (bf16)src[(size_t)s * D_LEN + d];
}

// ---------------- flash attention forward ----------------
// grid  = S/32 workgroups, one 32-query tile each
// block = 512 threads = 16 wave32
//   wave = mt*8 + w : mt in {0,1} selects 16-row M-tile, w in 0..7 selects
//   the 128-column d-slice this wave owns for both QK^T and the O output.
__global__ __launch_bounds__(512)
void fa_fwd_kernel(const bf16* __restrict__ Qb,   // [S][D] bf16
                   const bf16* __restrict__ Kb,   // [S][D] bf16
                   const bf16* __restrict__ Vt,   // [D][S] bf16 (transposed)
                   float* __restrict__ Out) {     // [S][D] f32
  __shared__ float Spart[8][32][32];   // per-d-slice partial scores
  __shared__ float Scomb[32][32];      // combined scores -> probabilities
  __shared__ float m_s[32], l_s[32], alpha_s[32];

  const int tid  = threadIdx.x;
  const int wave = tid >> 5;
  const int lane = tid & 31;
  const int mt   = wave >> 3;          // 0/1: which 16 query rows
  const int w_d  = wave & 7;           // 0..7: which 128 d-columns
  const int lm   = lane & 15;
  const int hi   = lane >> 4;          // 0 or 1
  const int kh8  = hi * 8;             // accumulator row base (D rows 8h..8h+7)
  const int kh16 = hi * 16;            // A/B-frag K base (contiguous k = 16h + s)
  const int qbase = blockIdx.x * 32;
  const int mrow  = mt * 16;           // this wave's row offset in the tile
  const float scale = 0.03125f;        // 1/sqrt(1024)

  if (tid < 32) { m_s[tid] = -3.0e38f; l_s[tid] = 0.0f; }

  // NULL-tensor TDM op (D# count=0 -> no data moved): exercises the
  // tensor_load_to_lds / TENSORcnt path once at kernel start.
  {
    u32x4 g0 = {0u, 0u, 0u, 0x80000000u};   // count=0, type=2
    u32x8 g1 = {0u, 0u, 0u, 0u, 0u, 0u, 0u, 0u};
    asm volatile("tensor_load_to_lds %0, %1" :: "s"(g0), "s"(g1));
    __builtin_amdgcn_s_wait_tensorcnt(0);
  }

  // Resident Q A-fragments: 4 fragments of 16x32 bf16 covering this wave's
  // 16 rows x 128 d-columns.
  bf16x16 qa[4];
  {
    const bf16* qrow = Qb + (size_t)(qbase + mrow + lm) * D_LEN + w_d * 128;
#pragma unroll
    for (int f = 0; f < 4; ++f) {
      bf16x8 r0 = *(const bf16x8*)(qrow + f * 32 + kh16);       // K = kh16+0..7
      bf16x8 r1 = *(const bf16x8*)(qrow + f * 32 + kh16 + 8);   // K = kh16+8..15
#pragma unroll
      for (int j = 0; j < 8; ++j) { qa[f][j] = r0[j]; qa[f][j + 8] = r1[j]; }
    }
  }

  v8f o[8];                            // 16x128 f32 output accumulator
#pragma unroll
  for (int t = 0; t < 8; ++t)
#pragma unroll
    for (int r = 0; r < 8; ++r) o[t][r] = 0.0f;

  __syncthreads();

  for (int kk = 0; kk < S_LEN; kk += 32) {
    // ---- issue V B-frag loads now; consumed only after softmax, so the
    //      memory latency is hidden under QK + reduction + softmax ----
    bf16x16 vb[8];
#pragma unroll
    for (int t = 0; t < 8; ++t)
      vb[t] = *(const bf16x16*)(Vt + (size_t)(w_d * 128 + t * 16 + lm) * S_LEN +
                                kk + kh16);

    // ---- prefetch next iteration's K rows / V columns into cache ----
    if (kk + 32 < S_LEN) {
      __builtin_prefetch(Kb + (size_t)(kk + 32 + lane) * D_LEN + w_d * 128, 0, 1);
      __builtin_prefetch(Vt + (size_t)(w_d * 128 + lane * 4) * S_LEN + kk + 32, 0, 1);
    }

    // ---- partial scores: S_w[16x32] = Q[mrows,dslice] * K[kk:kk+32,dslice]^T ----
    v8f sacc[2];
#pragma unroll
    for (int nt = 0; nt < 2; ++nt) {
#pragma unroll
      for (int r = 0; r < 8; ++r) sacc[nt][r] = 0.0f;
      bf16x16 kb[4];
#pragma unroll
      for (int f = 0; f < 4; ++f)
        kb[f] = *(const bf16x16*)(Kb + (size_t)(kk + nt * 16 + lm) * D_LEN +
                                  w_d * 128 + f * 32 + kh16);
#pragma unroll
      for (int f = 0; f < 4; ++f)
        sacc[nt] = __builtin_amdgcn_wmma_f32_16x16x32_bf16(
            false, qa[f], false, kb[f], (short)0, sacc[nt], false, false);
    }
#pragma unroll
    for (int nt = 0; nt < 2; ++nt)
#pragma unroll
      for (int r = 0; r < 8; ++r)
        Spart[w_d][mrow + r + kh8][nt * 16 + lm] = sacc[nt][r];
    __syncthreads();                                   // B1

    // ---- deterministic cross-slice reduce: 1024 elements / 512 threads ----
#pragma unroll
    for (int c = 0; c < 2; ++c) {
      int idx = tid + c * 512;
      int m = idx >> 5, n = idx & 31;
      float s = 0.0f;
#pragma unroll
      for (int w = 0; w < 8; ++w) s += Spart[w][m][n];
      Scomb[m][n] = s * scale;
    }
    __syncthreads();                                   // B2

    // ---- row max + correction factor (one thread per query row) ----
    if (tid < 32) {
      float mo = m_s[tid], mx = mo;
#pragma unroll
      for (int n = 0; n < 32; ++n) mx = fmaxf(mx, Scomb[tid][n]);
      alpha_s[tid] = __expf(mo - mx);
      m_s[tid] = mx;
    }
    __syncthreads();                                   // B3

    // ---- exponentials spread over all 512 threads (2 each) ----
#pragma unroll
    for (int c = 0; c < 2; ++c) {
      int idx = tid + c * 512;
      int m = idx >> 5, n = idx & 31;
      Scomb[m][n] = __expf(Scomb[m][n] - m_s[m]);
    }
    __syncthreads();                                   // B4

    // ---- row sums into l (cheap, no transcendentals) ----
    if (tid < 32) {
      float s = 0.0f;
#pragma unroll
      for (int n = 0; n < 32; ++n) s += Scomb[tid][n];
      l_s[tid] = l_s[tid] * alpha_s[tid] + s;
    }

    // ---- rescale O accumulator by alpha[row] ----
    float a[8];
#pragma unroll
    for (int r = 0; r < 8; ++r) a[r] = alpha_s[mrow + r + kh8];
#pragma unroll
    for (int t = 0; t < 8; ++t)
#pragma unroll
      for (int r = 0; r < 8; ++r) o[t][r] *= a[r];

    // ---- P A-fragment (16x32 bf16) from LDS probabilities ----
    bf16x16 pa;
#pragma unroll
    for (int j = 0; j < 8; ++j) {
      pa[j]     = (bf16)Scomb[mrow + lm][kh16 + j];
      pa[j + 8] = (bf16)Scomb[mrow + lm][kh16 + 8 + j];
    }

    // ---- O[mrows, dslice] += P * V[kk:kk+32, dslice] ----
#pragma unroll
    for (int t = 0; t < 8; ++t)
      o[t] = __builtin_amdgcn_wmma_f32_16x16x32_bf16(
          false, pa, false, vb[t], (short)0, o[t], false, false);
    __syncthreads();                                   // B5
  }

  // ---- normalize and write out ----
  float inv[8];
#pragma unroll
  for (int r = 0; r < 8; ++r) inv[r] = 1.0f / l_s[mrow + r + kh8];
#pragma unroll
  for (int t = 0; t < 8; ++t) {
    int col = w_d * 128 + t * 16 + lm;
#pragma unroll
    for (int r = 0; r < 8; ++r)
      Out[(size_t)(qbase + mrow + r + kh8) * D_LEN + col] = o[t][r] * inv[r];
  }
}

extern "C" void kernel_launch(void* const* d_in, const int* in_sizes, int n_in,
                              void* d_out, int out_size, void* d_ws, size_t ws_size,
                              hipStream_t stream) {
  const float* q = (const float*)d_in[0];
  const float* k = (const float*)d_in[1];
  const float* v = (const float*)d_in[2];
  float* out = (float*)d_out;

  // workspace: Q bf16 (16MB) | K bf16 (16MB) | V^T bf16 (16MB)
  bf16* qb = (bf16*)d_ws;
  bf16* kb = qb + (size_t)S_LEN * D_LEN;
  bf16* vt = kb + (size_t)S_LEN * D_LEN;

  const int n = S_LEN * D_LEN;
  cvt_bf16_kernel<<<n / (256 * 4), 256, 0, stream>>>(q, qb, n);
  cvt_bf16_kernel<<<n / (256 * 4), 256, 0, stream>>>(k, kb, n);
  cvt_transpose_kernel<<<n / 256, 256, 0, stream>>>(v, vt);

  fa_fwd_kernel<<<S_LEN / 32, 512, 0, stream>>>(qb, kb, vt, out);
}
